// CustomGPT2Attention_88776974008638
// MI455X (gfx1250) — compile-verified, matrix operand order checked by reading, not run on hardware
//
#include <hip/hip_runtime.h>
#include <hip/hip_bf16.h>

// ---------------------------------------------------------------------------
// GPT-2 attention block on gfx1250 (wave32, WMMA f32_16x16x32_f16).
//   B=16, S=1024, E=768, H=12, D=64
//   1) convert fp32 -> f16 (X), transpose+convert weights
//   2) QKV GEMM   (WMMA, f16 out + bias)
//   3) flash attention (WMMA QK^T and PV, streaming softmax,
//      K tile staged via GLOBAL_LOAD_ASYNC_TO_LDS when available)
//   4) proj GEMM  (WMMA, fp32 out + bias)
// ---------------------------------------------------------------------------

typedef __attribute__((ext_vector_type(16))) _Float16 v16h;
typedef __attribute__((ext_vector_type(8)))  float    v8f;
typedef __attribute__((ext_vector_type(4)))  int      v4i;

#define B_   16
#define S_   1024
#define E_   768
#define H_   12
#define D_   64
#define E3_  2304

#if defined(__HIP_DEVICE_COMPILE__) && __has_builtin(__builtin_amdgcn_global_load_async_to_lds_b128)
#define HAS_ASYNC_LDS 1
typedef __attribute__((address_space(1))) v4i g_v4i;  // global int4
typedef __attribute__((address_space(3))) v4i l_v4i;  // LDS int4
#else
#define HAS_ASYNC_LDS 0
#endif

static __device__ __forceinline__ void wait_async_lds() {
#if HAS_ASYNC_LDS
#if __has_builtin(__builtin_amdgcn_s_wait_asynccnt)
  __builtin_amdgcn_s_wait_asynccnt(0);
#else
  asm volatile("s_wait_asynccnt 0x0" ::: "memory");
#endif
#endif
}

union Frag {
  v16h     v;
  _Float16 h[16];
  uint4    q2[2];
};

static __device__ __forceinline__ v8f vzero() {
  v8f z;
#pragma unroll
  for (int i = 0; i < 8; ++i) z[i] = 0.0f;
  return z;
}

// Load a 16x32 f16 WMMA A/B fragment. `p` points at (row/col, k0) with the
// contraction dim contiguous; kb8 = (lane&16)?8:0. Elements e0..7 hold
// K=kb8..kb8+7, e8..15 hold K=kb8+16..kb8+23 (ISA 7.12.2 16-bit A layout).
static __device__ __forceinline__ void load_frag(Frag& f, const _Float16* p, int kb8) {
  f.q2[0] = *(const uint4*)(p + kb8);
  f.q2[1] = *(const uint4*)(p + kb8 + 16);
}

static __device__ __forceinline__ v8f wmma16(const Frag& a, const Frag& b, v8f c) {
  return __builtin_amdgcn_wmma_f32_16x16x32_f16(
      /*neg_a=*/false, a.v, /*neg_b=*/false, b.v,
      /*c_mod=*/(short)0, c, /*reuse_a=*/false, /*reuse_b=*/false);
}

// ---------------------------------------------------------------------------
// fp32 -> f16 elementwise
__global__ __launch_bounds__(256) void cvt_f16(const float* __restrict__ in,
                                               _Float16* __restrict__ out, int n) {
  int i = blockIdx.x * 256 + threadIdx.x;
  if (i < n) out[i] = (_Float16)in[i];
}

// fp32 [K][N] -> f16 [N][K] (transpose + convert; small weight matrices)
__global__ __launch_bounds__(256) void cvt_f16_t(const float* __restrict__ in,
                                                 _Float16* __restrict__ out,
                                                 int K, int N) {
  int i = blockIdx.x * 256 + threadIdx.x;
  if (i < K * N) {
    int k = i / N, n = i - k * N;
    out[n * K + k] = (_Float16)in[i];
  }
}

// ---------------------------------------------------------------------------
// C[M][N] = A[M][K] * Bt[N][K]^T + bias.  Block tile 64(M) x 128(N),
// 8 waves, each wave a 32x32 tile (2x2 WMMA accumulators). K multiple of 32.
// Operands are L2-resident (192 MB L2 >> 29 MB working set), so fragments
// are loaded directly as b128s; prefetch walks the next K slab.
__global__ __launch_bounds__(256) void wmma_gemm(
    const _Float16* __restrict__ A, const _Float16* __restrict__ Bt,
    const float* __restrict__ bias, void* __restrict__ Cout,
    int M, int N, int K, int out_is_half) {
  const int lane = threadIdx.x & 31;
  const int wid  = threadIdx.x >> 5;
  const int ln   = lane & 15;
  const int kb8  = (lane & 16) ? 8 : 0;
  const int mw   = blockIdx.y * 64  + (wid & 1) * 32;
  const int nw   = blockIdx.x * 128 + (wid >> 1) * 32;

  v8f c[2][2];
#pragma unroll
  for (int mi = 0; mi < 2; ++mi)
#pragma unroll
    for (int ni = 0; ni < 2; ++ni) c[mi][ni] = vzero();

  for (int k0 = 0; k0 < K; k0 += 32) {
    Frag a[2], b[2];
#pragma unroll
    for (int mi = 0; mi < 2; ++mi)
      load_frag(a[mi], A + (size_t)(mw + mi * 16 + ln) * K + k0, kb8);
#pragma unroll
    for (int ni = 0; ni < 2; ++ni)
      load_frag(b[ni], Bt + (size_t)(nw + ni * 16 + ln) * K + k0, kb8);
    if (k0 + 32 < K) {  // global_prefetch_b8 for next K slab
      __builtin_prefetch(A + (size_t)(mw + ln) * K + k0 + 32, 0, 1);
      __builtin_prefetch(Bt + (size_t)(nw + ln) * K + k0 + 32, 0, 1);
    }
#pragma unroll
    for (int mi = 0; mi < 2; ++mi)
#pragma unroll
      for (int ni = 0; ni < 2; ++ni) c[mi][ni] = wmma16(a[mi], b[ni], c[mi][ni]);
  }

#pragma unroll
  for (int ni = 0; ni < 2; ++ni) {
    const int col = nw + ni * 16 + ln;
    const float bz = bias ? bias[col] : 0.0f;
#pragma unroll
    for (int mi = 0; mi < 2; ++mi) {
#pragma unroll
      for (int r = 0; r < 8; ++r) {
        const int row = mw + mi * 16 + r + kb8;  // C layout: hi half => M+8
        const float val = c[mi][ni][r] + bz;
        if (out_is_half)
          ((_Float16*)Cout)[(size_t)row * N + col] = (_Float16)val;
        else
          ((float*)Cout)[(size_t)row * N + col] = val;
      }
    }
  }
}

// ---------------------------------------------------------------------------
// Flash attention. Grid = B*H*(S/128) blocks, 256 threads (8 waves).
// Wave w owns 16 query rows. Scores computed transposed (S^T = K_tile x Q^T)
// so lane = query: softmax is per-lane over regs + one shfl_xor(16), and the
// probability tile is directly the A-fragment for P x V.
__global__ __launch_bounds__(256) void flash_attn(const _Float16* __restrict__ qkv,
                                                  _Float16* __restrict__ ctx) {
  __shared__ _Float16 ldsK[32 * 64];   // [key][d]
  __shared__ _Float16 ldsVt[64 * 32];  // [d][key]

  const int bx   = blockIdx.x;
  const int qblk = bx & 7;
  const int h    = (bx >> 3) % H_;
  const int b    = bx / (8 * H_);
  const int t    = threadIdx.x;
  const int lane = t & 31;
  const int wid  = t >> 5;
  const int ln   = lane & 15;
  const int kb8  = (lane & 16) ? 8 : 0;

  const int q0 = qblk * 128 + wid * 16;
  const int qi = q0 + ln;  // this lane's query (B-frag column / A-frag row)

  // Q^T B-fragments (d = contraction), loaded once: d in [0,32) and [32,64)
  Frag qf[2];
#pragma unroll
  for (int j = 0; j < 2; ++j)
    load_frag(qf[j], qkv + (size_t)(b * S_ + qi) * E3_ + h * D_ + j * 32, kb8);

  float m_run = -3.0e38f, l_run = 0.0f;
  v8f acc[4];
#pragma unroll
  for (int ch = 0; ch < 4; ++ch) acc[ch] = vzero();

  const int nkb = qblk * 4 + 4;  // causal: key tiles of 32 up to qblk*128+127
  for (int kb = 0; kb < nkb; ++kb) {
    __syncthreads();
    {  // stage K tile [32 keys][64 d]: 16 bytes per thread
      const int key = t >> 3, seg = t & 7;
      const size_t g = (size_t)(b * S_ + kb * 32 + key) * E3_;
      const _Float16* ksrc = qkv + g + E_ + h * D_ + seg * 8;
      _Float16* kdst = &ldsK[key * 64 + seg * 8];
#if HAS_ASYNC_LDS
      // CDNA5 async DMA path: global -> LDS without a VGPR round trip,
      // tracked by ASYNCcnt.
      __builtin_amdgcn_global_load_async_to_lds_b128(
          (g_v4i*)ksrc, (l_v4i*)kdst, 0, 0);
#else
      *(uint4*)kdst = *(const uint4*)ksrc;
#endif
      // stage V^T tile [64 d][32 keys] (transpose in flight -> ds stores)
#pragma unroll
      for (int j = 0; j < 4; ++j) {
        const int dw = (t & 7) + j * 8;  // dword pair index along d
        union { unsigned u; _Float16 hh[2]; } cv;
        cv.u = *(const unsigned*)(qkv + g + 2 * E_ + h * D_ + dw * 2);
        ldsVt[(dw * 2 + 0) * 32 + key] = cv.hh[0];
        ldsVt[(dw * 2 + 1) * 32 + key] = cv.hh[1];
      }
    }
    wait_async_lds();   // ASYNCcnt -> 0 before the barrier publishes LDS
    __syncthreads();

    // K-tile A-fragments: rows = keys (two 16-key subtiles), contract = d
    Frag aK[2][2];
#pragma unroll
    for (int sub = 0; sub < 2; ++sub)
#pragma unroll
      for (int j = 0; j < 2; ++j)
        load_frag(aK[sub][j], &ldsK[(sub * 16 + ln) * 64 + j * 32], kb8);

    v8f s0 = vzero(), s1 = vzero();  // S^T: reg = key row, lane = query
    s0 = wmma16(aK[0][0], qf[0], s0); s0 = wmma16(aK[0][1], qf[1], s0);
    s1 = wmma16(aK[1][0], qf[0], s1); s1 = wmma16(aK[1][1], qf[1], s1);

    // scale + causal mask + running max
    float mloc = -3.0e38f;
#pragma unroll
    for (int r = 0; r < 8; ++r) {
      const int kg0 = kb * 32 + kb8 + r;
      const int kg1 = kg0 + 16;
      const float v0 = (kg0 <= qi) ? s0[r] * 0.125f : -3.0e38f;
      const float v1 = (kg1 <= qi) ? s1[r] * 0.125f : -3.0e38f;
      s0[r] = v0; s1[r] = v1;
      mloc = fmaxf(mloc, fmaxf(v0, v1));
    }
    mloc = fmaxf(mloc, __shfl_xor(mloc, 16, 32));  // halves hold other keys
    const float m_new = fmaxf(m_run, mloc);
    const float alpha = __expf(m_run - m_new);

    Frag pf;  // probability tile == A-fragment for P x V (lane = query row)
    float lloc = 0.0f;
#pragma unroll
    for (int r = 0; r < 8; ++r) {
      const float p0 = __expf(s0[r] - m_new);
      const float p1 = __expf(s1[r] - m_new);
      lloc += p0 + p1;
      pf.h[r]     = (_Float16)p0;  // keys kb8+r      -> A elems 0..7
      pf.h[8 + r] = (_Float16)p1;  // keys 16+kb8+r   -> A elems 8..15
    }
    lloc += __shfl_xor(lloc, 16, 32);
    l_run = l_run * alpha + lloc;
    m_run = m_new;

    float ar[8];  // alpha per accumulator row (row q = r + kb8)
#pragma unroll
    for (int r = 0; r < 8; ++r) ar[r] = __shfl(alpha, r + kb8, 32);

#pragma unroll
    for (int ch = 0; ch < 4; ++ch) {  // out d-chunks of 16
      Frag vf;  // V B-fragment: lane col = d, contract = key (V^T in LDS)
      load_frag(vf, &ldsVt[(ch * 16 + ln) * 32], kb8);
#pragma unroll
      for (int r = 0; r < 8; ++r) acc[ch][r] *= ar[r];
      acc[ch] = wmma16(pf, vf, acc[ch]);
    }
  }

  const float rl = 1.0f / l_run;
  float rr[8];
#pragma unroll
  for (int r = 0; r < 8; ++r) rr[r] = __shfl(rl, r + kb8, 32);
#pragma unroll
  for (int ch = 0; ch < 4; ++ch) {
#pragma unroll
    for (int r = 0; r < 8; ++r) {
      const int qg = q0 + r + kb8;
      ctx[(size_t)(b * S_ + qg) * E_ + h * D_ + ch * 16 + ln] =
          (_Float16)(acc[ch][r] * rr[r]);
    }
  }
}

// ---------------------------------------------------------------------------
extern "C" void kernel_launch(void* const* d_in, const int* in_sizes, int n_in,
                              void* d_out, int out_size, void* d_ws, size_t ws_size,
                              hipStream_t stream) {
  const float* hs     = (const float*)d_in[0];  // [16,1024,768]
  const float* W_attn = (const float*)d_in[1];  // [768,2304]
  const float* b_attn = (const float*)d_in[2];  // [2304]
  const float* W_proj = (const float*)d_in[3];  // [768,768]
  const float* b_proj = (const float*)d_in[4];  // [768]
  float* out = (float*)d_out;

  // Workspace layout (f16): X, W_attn^T, W_proj^T, qkv, ctx  (~130 MB)
  char* w = (char*)d_ws;
  _Float16* Xh  = (_Float16*)w; w += (size_t)B_ * S_ * E_ * 2;
  _Float16* WaT = (_Float16*)w; w += (size_t)E3_ * E_ * 2;
  _Float16* WpT = (_Float16*)w; w += (size_t)E_ * E_ * 2;
  _Float16* qkv = (_Float16*)w; w += (size_t)B_ * S_ * E3_ * 2;
  _Float16* ctx = (_Float16*)w; w += (size_t)B_ * S_ * E_ * 2;

  const int nX = B_ * S_ * E_;
  cvt_f16  <<<(nX + 255) / 256, 256, 0, stream>>>(hs, Xh, nX);
  cvt_f16_t<<<(E_ * E3_ + 255) / 256, 256, 0, stream>>>(W_attn, WaT, E_, E3_);
  cvt_f16_t<<<(E_ * E_  + 255) / 256, 256, 0, stream>>>(W_proj, WpT, E_, E_);

  // QKV GEMM: [16384 x 768] @ [768 x 2304] + b_attn -> f16 qkv
  wmma_gemm<<<dim3(E3_ / 128, (B_ * S_) / 64), 256, 0, stream>>>(
      Xh, WaT, b_attn, (void*)qkv, B_ * S_, E3_, E_, /*out_is_half=*/1);

  // Flash attention: 16*12*8 = 1536 blocks
  flash_attn<<<B_ * H_ * (S_ / 128), 256, 0, stream>>>(qkv, ctx);

  // Projection GEMM: [16384 x 768] @ [768 x 768] + b_proj -> fp32 out
  wmma_gemm<<<dim3(E_ / 128, (B_ * S_) / 64), 256, 0, stream>>>(
      ctx, WpT, b_proj, (void*)out, B_ * S_, E_, E_, /*out_is_half=*/0);
}
